// RGLRU_20005957665326
// MI455X (gfx1250) — matrix operand order repaired and checked by reading, not compiled, not run on hardware
//
#include <hip/hip_runtime.h>
#include <math.h>

typedef __attribute__((ext_vector_type(16))) __bf16 v16bf;
typedef __attribute__((ext_vector_type(8)))  float  v8f;

#define BB   8
#define TT   4096
#define DHID 2048
#define DH2  1024
#define NCH  32
#define CLEN 128

// LDS tile row stride in bf16 elements: 80B rows -> 16B aligned for b128,
// 20-bank stride on the 64-bank LDS (avoids the 16-bank 4-way conflict).
#define LDS_STR 40

// ---------- fp32 -> bf16 (round to nearest even) ----------
static __device__ __forceinline__ unsigned short f2bf(float f) {
  union { float f; unsigned u; } v; v.f = f;
  unsigned u = v.u;
  unsigned r = (u + 0x7FFFu + ((u >> 16) & 1u)) >> 16;
  return (unsigned short)r;
}

__global__ void cvt_f32_bf16(const float* __restrict__ in,
                             unsigned short* __restrict__ out, size_t n) {
  size_t i = (size_t)blockIdx.x * blockDim.x + threadIdx.x;
  size_t stride = (size_t)gridDim.x * blockDim.x;
  for (; i < n; i += stride) out[i] = f2bf(in[i]);
}

// ---------- fp32 [K][N] -> bf16 transposed [N][K], batched ----------
__global__ void cvt_transpose_bf16(const float* __restrict__ in,
                                   unsigned short* __restrict__ out,
                                   int K, int N, int nb) {
  size_t total = (size_t)nb * K * N;
  size_t mat = (size_t)K * N;
  size_t i = (size_t)blockIdx.x * blockDim.x + threadIdx.x;
  size_t stride = (size_t)gridDim.x * blockDim.x;
  for (; i < total; i += stride) {
    size_t b = i / mat, rem = i % mat;
    size_t n = rem / K, k = rem % K;
    out[i] = f2bf(in[b * mat + k * N + n]);   // coalesced writes
  }
}

// ---------- A-fragment K mapping for 16-bit A 16x32 (ISA 7.12.2) ----------
// For fixed lane-group g this yields two contiguous 8-element (16B) runs:
// [8g .. 8g+7] and [16+8g .. 23+8g]  -> compiler emits 2x ds_load_b128.
static __device__ __forceinline__ int kmapA(int e, int g) {
  int v = e >> 1, p = e & 1;
  return ((v >> 2) << 4) + (g << 3) + ((v & 3) << 1) + p;
}

// ---------- generic bf16 WMMA GEMM: C = act(A*Bt^T + bias) ----------
// A:   row-major [M][K] bf16 activations.
// Btg: row-major [N][K] bf16 (PRE-TRANSPOSED weights), so both tile fills
//      are identical wide row-major copies (global_load_b128 -> ds_store_b128).
// block tile 128x128, 256 threads = 8 wave32 (4 along M x 2 along N),
// wave tile 32x64 = 2x4 fragments of 16x16, K step 32.
// Register-staged pipeline: next tile's global loads issue before the WMMA
// section; their waitcnt lands just before next iteration's ds_stores.
__global__ __launch_bounds__(256) void gemm_bf16(
    const unsigned short* __restrict__ A, const unsigned short* __restrict__ Btg,
    const float* __restrict__ bias, float* __restrict__ C,
    unsigned short* __restrict__ Cb,
    int K, int lda, int ldbt, int ldc, int act)
{
  __shared__ __align__(16) unsigned short As[128 * LDS_STR];
  __shared__ __align__(16) unsigned short Bs[128 * LDS_STR];

  const int tid  = threadIdx.x;
  const int lane = tid & 31;
  const int wave = tid >> 5;
  const int g    = lane >> 4;
  const int ln   = lane & 15;
  const int wm   = wave & 3;   // wave row (M)
  const int wn   = wave >> 2;  // wave col (N)
  const int rowBase = blockIdx.x * 128;
  const int colBase = blockIdx.y * 128;

  // fill indices: 128 rows x 32 cols bf16 = 512 x 16B chunks, 2 per thread
  const int fr0 = (tid + 0)   >> 2, fq0 = (tid + 0)   & 3;
  const int fr1 = (tid + 256) >> 2, fq1 = (tid + 256) & 3;
  const unsigned short* Ar0 = A   + (size_t)(rowBase + fr0) * lda  + fq0 * 8;
  const unsigned short* Ar1 = A   + (size_t)(rowBase + fr1) * lda  + fq1 * 8;
  const unsigned short* Br0 = Btg + (size_t)(colBase + fr0) * ldbt + fq0 * 8;
  const unsigned short* Br1 = Btg + (size_t)(colBase + fr1) * ldbt + fq1 * 8;

  v8f acc[2][4];
  const v8f zero = {0.f,0.f,0.f,0.f,0.f,0.f,0.f,0.f};
  for (int i = 0; i < 2; i++)
    for (int j = 0; j < 4; j++) acc[i][j] = zero;

  // prologue: stage tile k0=0 in registers
  uint4 ra0 = *(const uint4*)(Ar0);
  uint4 ra1 = *(const uint4*)(Ar1);
  uint4 rb0 = *(const uint4*)(Br0);
  uint4 rb1 = *(const uint4*)(Br1);

  for (int k0 = 0; k0 < K; k0 += 32) {
    __syncthreads();                       // all waves done reading prev tile
    *(uint4*)(&As[fr0 * LDS_STR + fq0 * 8]) = ra0;
    *(uint4*)(&As[fr1 * LDS_STR + fq1 * 8]) = ra1;
    *(uint4*)(&Bs[fr0 * LDS_STR + fq0 * 8]) = rb0;
    *(uint4*)(&Bs[fr1 * LDS_STR + fq1 * 8]) = rb1;
    __syncthreads();                       // tile visible to all waves

    // issue next tile's global loads now; they complete under the WMMAs
    int kn = (k0 + 32 < K) ? (k0 + 32) : k0;   // clamped (uniform, branchless)
    ra0 = *(const uint4*)(Ar0 + kn);
    ra1 = *(const uint4*)(Ar1 + kn);
    rb0 = *(const uint4*)(Br0 + kn);
    rb1 = *(const uint4*)(Br1 + kn);
    // prefetch two tiles ahead to keep HBM->L2 ahead of the pipeline
    int kp = (k0 + 64 < K) ? (k0 + 64) : kn;
    __builtin_prefetch((const char*)(A   + (size_t)(rowBase + (tid >> 1)) * lda  + kp), 0, 0);
    __builtin_prefetch((const char*)(Btg + (size_t)(colBase + (tid >> 1)) * ldbt + kp), 0, 0);

    const __bf16* Asb = (const __bf16*)As;
    const __bf16* Bsb = (const __bf16*)Bs;
    v16bf afr[2], bfr[4];
    #pragma unroll
    for (int fm = 0; fm < 2; fm++) {
      int row = wm * 32 + fm * 16 + ln;
      #pragma unroll
      for (int e = 0; e < 16; e++) afr[fm][e] = Asb[row * LDS_STR + kmapA(e, g)];
    }
    #pragma unroll
    for (int fn = 0; fn < 4; fn++) {
      int col = wn * 64 + fn * 16 + ln;
      #pragma unroll
      for (int e = 0; e < 16; e++) bfr[fn][e] = Bsb[col * LDS_STR + kmapA(e, g)];
    }
    #pragma unroll
    for (int fm = 0; fm < 2; fm++)
      #pragma unroll
      for (int fn = 0; fn < 4; fn++)
        acc[fm][fn] = __builtin_amdgcn_wmma_f32_16x16x32_bf16(
            false, afr[fm], false, bfr[fn], (short)0, acc[fm][fn], false, false);
  }

  // epilogue: C/D layout -> row = r + 8*g, col = lane%16 (ISA 7.12.2)
  #pragma unroll
  for (int fm = 0; fm < 2; fm++) {
    #pragma unroll
    for (int fn = 0; fn < 4; fn++) {
      int col = colBase + wn * 64 + fn * 16 + ln;
      float bv = bias ? bias[col] : 0.f;
      #pragma unroll
      for (int r = 0; r < 8; r++) {
        int row = rowBase + wm * 32 + fm * 16 + g * 8 + r;
        float v = acc[fm][fn][r] + bv;
        if (act) v = 1.f / (1.f + expf(-v));
        C[(size_t)row * ldc + col] = v;
        if (Cb) Cb[(size_t)row * ldc + col] = f2bf(v);
      }
    }
  }
}

// ---------- gate math helper ----------
struct GateVals { float are, aim, xcr, xci; };

static __device__ __forceinline__ GateVals gate_step(
    float gav, float gxv, float xr, float xi, float sp, float ai)
{
  float lre = -8.f * gav * sp;       // log|a|
  float lim = ai * gav;              // arg(a)
  float ea  = expf(lre);
  float c1  = cosf(lim), s1 = sinf(lim);
  float are = ea * c1, aim = ea * s1;
  float e2  = ea * ea;
  float c2v = cosf(2.f * lim), s2v = sinf(2.f * lim);
  float wr  = 1.f - e2 * c2v;        // 1 - a^2 (complex)
  float wi  = -e2 * s2v;
  float rm  = sqrtf(wr * wr + wi * wi);
  float sre = sqrtf(fmaxf(0.f, 0.5f * (rm + wr)));
  float sim = copysignf(sqrtf(fmaxf(0.f, 0.5f * (rm - wr))), wi);
  float bxr = gxv * xr, bxi = gxv * xi;
  GateVals o;
  o.are = are; o.aim = aim;
  o.xcr = sre * bxr - sim * bxi;
  o.xci = sre * bxi + sim * bxr;
  return o;
}

// ---------- pass 1: per-chunk scan summaries (A_chunk, H_chunk) ----------
__global__ void pass1_chunk(const float* __restrict__ xh, const float* __restrict__ gx,
                            const float* __restrict__ ga, const float* __restrict__ ar_p,
                            const float* __restrict__ ai_p,
                            float2* __restrict__ Ach, float2* __restrict__ Hch)
{
  int c  = blockIdx.x * blockDim.x + threadIdx.x;   // channel 0..1023
  int ch = blockIdx.y;                              // chunk
  int b  = blockIdx.z;                              // batch
  float ar = ar_p[c], ai = ai_p[c];
  float sp = (ar > 20.f) ? ar : log1pf(expf(ar));   // softplus
  float Ar = 1.f, Ai = 0.f, hr = 0.f, hi = 0.f;
  for (int s = 0; s < CLEN; s++) {
    size_t i2 = (size_t)b * TT + (size_t)ch * CLEN + s;
    GateVals gv = gate_step(ga[i2 * DH2 + c], gx[i2 * DH2 + c],
                            xh[i2 * DHID + c], xh[i2 * DHID + DH2 + c], sp, ai);
    float nhr = gv.are * hr - gv.aim * hi + gv.xcr;
    float nhi = gv.are * hi + gv.aim * hr + gv.xci;
    hr = nhr; hi = nhi;
    float nAr = gv.are * Ar - gv.aim * Ai;
    float nAi = gv.are * Ai + gv.aim * Ar;
    Ar = nAr; Ai = nAi;
  }
  size_t o = ((size_t)b * NCH + ch) * DH2 + c;
  Ach[o] = make_float2(Ar, Ai);
  Hch[o] = make_float2(hr, hi);
}

// ---------- pass 2: sequential scan across chunk summaries ----------
__global__ void pass2_scan(const float2* __restrict__ Ach, const float2* __restrict__ Hch,
                           float2* __restrict__ Sin)
{
  int i = blockIdx.x * blockDim.x + threadIdx.x;    // 0..8191
  int b = i >> 10, c = i & 1023;
  float cr = 0.f, ci = 0.f;
  for (int ch = 0; ch < NCH; ch++) {
    size_t o = ((size_t)b * NCH + ch) * DH2 + c;
    Sin[o] = make_float2(cr, ci);                   // incoming state for this chunk
    float2 Av = Ach[o], Hv = Hch[o];
    float nr = Av.x * cr - Av.y * ci + Hv.x;
    float ni = Av.x * ci + Av.y * cr + Hv.y;
    cr = nr; ci = ni;
  }
}

// ---------- pass 3: re-scan seeded with chunk-incoming state, emit h_m (bf16) + h_last ----------
__global__ void pass3_apply(const float* __restrict__ xh, const float* __restrict__ gx,
                            const float* __restrict__ ga, const float* __restrict__ ar_p,
                            const float* __restrict__ ai_p, const float2* __restrict__ Sin,
                            unsigned short* __restrict__ hm_bf, float* __restrict__ out)
{
  int c  = blockIdx.x * blockDim.x + threadIdx.x;
  int ch = blockIdx.y;
  int b  = blockIdx.z;
  float ar = ar_p[c], ai = ai_p[c];
  float sp = (ar > 20.f) ? ar : log1pf(expf(ar));
  float2 seed = Sin[((size_t)b * NCH + ch) * DH2 + c];
  float hr = seed.x, hi = seed.y;
  for (int s = 0; s < CLEN; s++) {
    size_t i2 = (size_t)b * TT + (size_t)ch * CLEN + s;
    GateVals gv = gate_step(ga[i2 * DH2 + c], gx[i2 * DH2 + c],
                            xh[i2 * DHID + c], xh[i2 * DHID + DH2 + c], sp, ai);
    float nhr = gv.are * hr - gv.aim * hi + gv.xcr;
    float nhi = gv.are * hi + gv.aim * hr + gv.xci;
    hr = nhr; hi = nhi;
    hm_bf[i2 * DHID + c]       = f2bf(hr);
    hm_bf[i2 * DHID + DH2 + c] = f2bf(hi);
  }
  if (ch == NCH - 1) {  // h_last = h_m[:, T-1]
    size_t tail = (size_t)BB * TT * DHID;
    out[tail + (size_t)b * DHID + c]       = hr;
    out[tail + (size_t)b * DHID + DH2 + c] = hi;
  }
}

extern "C" void kernel_launch(void* const* d_in, const int* in_sizes, int n_in,
                              void* d_out, int out_size, void* d_ws, size_t ws_size,
                              hipStream_t stream)
{
  const float* x      = (const float*)d_in[0];
  const float* W_in   = (const float*)d_in[1];
  const float* b_in   = (const float*)d_in[2];
  const float* W_gx   = (const float*)d_in[3];
  const float* b_gx   = (const float*)d_in[4];
  const float* W_ga   = (const float*)d_in[5];
  const float* b_ga   = (const float*)d_in[6];
  const float* a_real = (const float*)d_in[7];
  const float* a_imag = (const float*)d_in[8];
  const float* W_out  = (const float*)d_in[9];
  const float* b_out  = (const float*)d_in[10];
  float* out = (float*)d_out;

  char* ws = (char*)d_ws;
  size_t off = 0;
  auto alloc = [&](size_t bytes) -> char* {
    char* p = ws + off; off += (bytes + 255) & ~(size_t)255; return p;
  };
  const size_t M = (size_t)BB * TT;  // 32768 rows

  unsigned short* x_bf     = (unsigned short*)alloc(M * DHID * 2);
  unsigned short* WinT_bf  = (unsigned short*)alloc((size_t)DHID * DHID * 2);
  unsigned short* WgxT_bf  = (unsigned short*)alloc((size_t)8 * 256 * 128 * 2);
  unsigned short* WgaT_bf  = (unsigned short*)alloc((size_t)8 * 256 * 128 * 2);
  unsigned short* WoutT_bf = (unsigned short*)alloc((size_t)DHID * DHID * 2);
  float*          xh       = (float*)alloc(M * DHID * 4);
  unsigned short* xh_bf    = (unsigned short*)alloc(M * DHID * 2);
  float*          gate_x   = (float*)alloc(M * DH2 * 4);
  float*          gate_a   = (float*)alloc(M * DH2 * 4);
  unsigned short* hm_bf    = (unsigned short*)alloc(M * DHID * 2);
  float2*         Ach      = (float2*)alloc((size_t)BB * NCH * DH2 * 8);
  float2*         Hch      = (float2*)alloc((size_t)BB * NCH * DH2 * 8);
  float2*         Sin      = (float2*)alloc((size_t)BB * NCH * DH2 * 8);

  // activations: plain fp32 -> bf16; weights: fp32 -> bf16 TRANSPOSED ([N][K])
  cvt_f32_bf16<<<4096, 256, 0, stream>>>(x, x_bf, M * DHID);
  cvt_transpose_bf16<<<4096, 256, 0, stream>>>(W_in,  WinT_bf,  DHID, DHID, 1);
  cvt_transpose_bf16<<<1024, 256, 0, stream>>>(W_gx,  WgxT_bf,  256, 128, 8);
  cvt_transpose_bf16<<<1024, 256, 0, stream>>>(W_ga,  WgaT_bf,  256, 128, 8);
  cvt_transpose_bf16<<<4096, 256, 0, stream>>>(W_out, WoutT_bf, DHID, DHID, 1);

  // GEMM1: xh = x @ W_in + b_in   (f32 + bf16 mirror)
  gemm_bf16<<<dim3(256, 16), 256, 0, stream>>>(
      x_bf, WinT_bf, b_in, xh, xh_bf, DHID, DHID, DHID, DHID, 0);

  // gate GEMMs: block-diagonal, sigmoid fused (transposed block = [128][256])
  for (int h = 0; h < 8; h++) {
    gemm_bf16<<<dim3(256, 1), 256, 0, stream>>>(
        xh_bf + h * 256, WgxT_bf + (size_t)h * 128 * 256, b_gx + h * 128,
        gate_x + h * 128, nullptr, 256, DHID, 256, DH2, 1);
    gemm_bf16<<<dim3(256, 1), 256, 0, stream>>>(
        xh_bf + h * 256, WgaT_bf + (size_t)h * 128 * 256, b_ga + h * 128,
        gate_a + h * 128, nullptr, 256, DHID, 256, DH2, 1);
  }

  // chunked complex linear scan
  pass1_chunk<<<dim3(4, NCH, BB), 256, 0, stream>>>(xh, gate_x, gate_a, a_real, a_imag, Ach, Hch);
  pass2_scan <<<32, 256, 0, stream>>>(Ach, Hch, Sin);
  pass3_apply<<<dim3(4, NCH, BB), 256, 0, stream>>>(xh, gate_x, gate_a, a_real, a_imag, Sin, hm_bf, out);

  // GEMM3: out = h_m @ W_out + b_out
  gemm_bf16<<<dim3(256, 16), 256, 0, stream>>>(
      hm_bf, WoutT_bf, b_out, out, nullptr, DHID, DHID, DHID, DHID, 0);
}
